// AFM_15487652069575
// MI455X (gfx1250) — matrix operand order repaired and checked by reading, not compiled, not run on hardware
//
#include <hip/hip_runtime.h>

#define BATCH     8192
#define NF        30
#define NPAIR     435      // C(30,2)
#define NPT       28       // ceil(435/16) pair tiles
#define PPAD      448      // 28*16
#define DIM       64
#define ATT       32
#define TPITCH    20       // sT row pitch in floats: 80B (16B-aligned, bank-spread)

typedef __attribute__((ext_vector_type(16))) _Float16 v16h;
typedef __attribute__((ext_vector_type(8)))  _Float16 v8h;
typedef __attribute__((ext_vector_type(4)))  _Float16 v4h;
typedef __attribute__((ext_vector_type(2)))  _Float16 v2h;
typedef __attribute__((ext_vector_type(8)))  float    v8f;
typedef __attribute__((ext_vector_type(4)))  float    v4f;

__global__ __launch_bounds__(256) void afm_wmma_kernel(
    const int*   __restrict__ x,      // [B, NF]
    const float* __restrict__ emb,    // [VOCAB, DIM]
    const float* __restrict__ W,      // [DIM, ATT]
    const float* __restrict__ attb,   // [ATT]
    const float* __restrict__ hvec,   // [ATT]
    const float* __restrict__ d1w,    // [32, 64]
    const float* __restrict__ d1b,    // [32]
    const float* __restrict__ d2w,    // [16, 32]
    const float* __restrict__ d2b,    // [16]
    const float* __restrict__ fw,     // [1, 16]
    const float* __restrict__ fb,     // [1]
    float*       __restrict__ out)    // [B, 1]
{
    // Per-wave LDS slices (8 waves / block) -- ~59 KB total
    __shared__ alignas(16) _Float16 sE[8][NF * DIM];      // 30720 B f16 embeddings
    __shared__ alignas(16) float sAtt[8][PPAD];           // 14336 B logits -> att
    __shared__ alignas(16) float sT[8][16 * TPITCH];      // 10240 B transpose-reduce
    __shared__ alignas(16) float sPool[8][DIM];           //  2048 B
    __shared__ alignas(16) float sM1[8][ATT];             //  1024 B
    __shared__ unsigned int sRC[PPAD];                    //  1792 B packed r*DIM | c*DIM<<16

    const int tid  = threadIdx.x;
    const int wave = tid >> 5;
    const int lane = tid & 31;
    const int l16  = lane & 15;
    const int kh   = lane >> 4;

    if (tid == 0) {
        int p = 0;
        for (int r = 0; r < NF; ++r)
            for (int c = r + 1; c < NF; ++c) {
                sRC[p++] = (unsigned)(r * DIM) | ((unsigned)(c * DIM) << 16);
            }
        for (; p < PPAD; ++p) sRC[p] = 0u | ((unsigned)DIM << 16);
    }
    __syncthreads();

    // ---- Resident B fragments from W (64x32), built once per wave ----
    // B layout (f16 32x16): slot h<8 -> k = kh*8+h ; h>=8 -> k = 16+kh*8+(h-8)
    v16h b00, b01, b10, b11;
    #pragma unroll
    for (int h = 0; h < 16; ++h) {
        const int k = (h < 8) ? (kh * 8 + h) : (16 + kh * 8 + (h - 8));
        b00[h] = (_Float16)W[(k     ) * ATT + l16     ];
        b01[h] = (_Float16)W[(k     ) * ATT + l16 + 16];
        b10[h] = (_Float16)W[(k + 32) * ATT + l16     ];
        b11[h] = (_Float16)W[(k + 32) * ATT + l16 + 16];
    }
    const float hA    = hvec[l16];
    const float hB    = hvec[l16 + 16];
    const float biasA = attb[l16];
    const float biasB = attb[l16 + 16];

    const int gwave  = blockIdx.x * 8 + wave;
    const int stride = gridDim.x * 8;

    for (int b = gwave; b < BATCH; b += stride) {
        // ---- 1. gather embeddings -> LDS f16 (half-wave per field row, b128 loads) ----
        #pragma unroll 3
        for (int i = 0; i < NF / 2; ++i) {
            const int f   = 2 * i + kh;
            const int row = x[b * NF + f];
            const v4f v = *(const v4f*)(emb + (size_t)row * DIM + 4 * l16);
            *(v4h*)&sE[wave][f * DIM + 4 * l16] = __builtin_convertvector(v, v4h);
        }
        __syncthreads();

        // ---- 2. attention logits: relu(hb @ W + b) . h  via WMMA f16 ----
        for (int pt = 0; pt < NPT; ++pt) {
            const unsigned rc = sRC[pt * 16 + l16];
            const int roff = (int)(rc & 0xffffu);
            const int coff = (int)(rc >> 16);
            // A layout (f16 16x32): h<8 -> k = kh*8+h ; h>=8 -> k = 16+kh*8+(h-8)
            const int base = kh * 8;
            v8h rlo = *(const v8h*)&sE[wave][roff + base];
            v8h rhi = *(const v8h*)&sE[wave][roff + base + 16];
            v8h clo = *(const v8h*)&sE[wave][coff + base];
            v8h chi = *(const v8h*)&sE[wave][coff + base + 16];
            v8h plo = rlo * clo;                      // v_pk_mul_f16
            v8h phi = rhi * chi;
            v16h a0 = __builtin_shufflevector(plo, phi, 0,1,2,3,4,5,6,7,8,9,10,11,12,13,14,15);
            rlo = *(const v8h*)&sE[wave][roff + 32 + base];
            rhi = *(const v8h*)&sE[wave][roff + 32 + base + 16];
            clo = *(const v8h*)&sE[wave][coff + 32 + base];
            chi = *(const v8h*)&sE[wave][coff + 32 + base + 16];
            plo = rlo * clo;
            phi = rhi * chi;
            v16h a1 = __builtin_shufflevector(plo, phi, 0,1,2,3,4,5,6,7,8,9,10,11,12,13,14,15);

            v8f c0 = {};   // N = 0..15
            v8f c1 = {};   // N = 16..31
            c0 = __builtin_amdgcn_wmma_f32_16x16x32_f16(false, a0, false, b00, (short)0, c0, false, false);
            c0 = __builtin_amdgcn_wmma_f32_16x16x32_f16(false, a1, false, b10, (short)0, c0, false, false);
            c1 = __builtin_amdgcn_wmma_f32_16x16x32_f16(false, a0, false, b01, (short)0, c1, false, false);
            c1 = __builtin_amdgcn_wmma_f32_16x16x32_f16(false, a1, false, b11, (short)0, c1, false, false);

            // relu(+bias) and h-weight, then LDS transpose-reduce over N
            #pragma unroll
            for (int vv = 0; vv < 8; ++vv) {
                const float t = fmaxf(c0[vv] + biasA, 0.f) * hA
                              + fmaxf(c1[vv] + biasB, 0.f) * hB;
                sT[wave][(kh * 8 + vv) * TPITCH + l16] = t;   // sT[m][n]
            }
            __builtin_amdgcn_fence(__ATOMIC_RELEASE, "wavefront");
            __builtin_amdgcn_wave_barrier();
            __builtin_amdgcn_fence(__ATOMIC_ACQUIRE, "wavefront");
            if (lane < 16) {
                const v4f* rowp = (const v4f*)&sT[wave][lane * TPITCH];
                const v4f s4 = rowp[0] + rowp[1] + rowp[2] + rowp[3];
                const float s = s4[0] + s4[1] + s4[2] + s4[3];
                const int pm = pt * 16 + lane;
                sAtt[wave][pm] = (pm < NPAIR) ? s : -1e30f;
            }
            __builtin_amdgcn_fence(__ATOMIC_RELEASE, "wavefront");
            __builtin_amdgcn_wave_barrier();   // keep next tile's sT stores after the reads
        }
        __syncthreads();

        // ---- 3. softmax over the 435 pair logits (wave-local) ----
        float mx = -1e30f;
        for (int p = lane; p < NPAIR; p += 32) mx = fmaxf(mx, sAtt[wave][p]);
        for (int s = 16; s > 0; s >>= 1) mx = fmaxf(mx, __shfl_xor(mx, s));
        float sum = 0.f;
        for (int p = lane; p < PPAD; p += 32) {
            const float e2 = (p < NPAIR) ? __expf(sAtt[wave][p] - mx) : 0.f;
            sAtt[wave][p] = e2;
            sum += e2;
        }
        for (int s = 16; s > 0; s >>= 1) sum += __shfl_xor(sum, s);
        const float inv = 1.f / sum;
        __syncthreads();

        // ---- 4. pooled[d] = sum_p att[p] * e[r][d]*e[c][d]  (lane owns d=2*lane,2*lane+1) ----
        const int dd = 2 * lane;
        float acc0 = 0.f, acc1 = 0.f;
        #pragma unroll 5
        for (int p = 0; p < NPAIR; ++p) {
            const unsigned rc = sRC[p];                 // broadcast
            const float w = sAtt[wave][p];              // broadcast
            const v2h er = *(const v2h*)&sE[wave][(rc & 0xffffu) + dd];
            const v2h ec = *(const v2h*)&sE[wave][(rc >> 16) + dd];
            const v2h pr = er * ec;                     // v_pk_mul_f16
            acc0 += w * (float)pr[0];
            acc1 += w * (float)pr[1];
        }
        {
            float2 pw; pw.x = acc0 * inv; pw.y = acc1 * inv;
            *(float2*)&sPool[wave][dd] = pw;
        }
        __syncthreads();

        // ---- 5. MLP head: 64 -> 32 -> 16 -> 1 (sigmoid) ----
        {
            const v4f* wrow = (const v4f*)(d1w + lane * DIM);
            const v4f* pool = (const v4f*)&sPool[wave][0];
            v4f acc = {0.f, 0.f, 0.f, 0.f};
            #pragma unroll
            for (int i = 0; i < DIM / 4; ++i) acc += wrow[i] * pool[i];
            const float o1 = d1b[lane] + acc[0] + acc[1] + acc[2] + acc[3];
            sM1[wave][lane] = fmaxf(o1, 0.f);
        }
        __syncthreads();

        float red = 0.f;
        if (lane < 16) {
            const v4f* wrow = (const v4f*)(d2w + lane * ATT);
            const v4f* m1   = (const v4f*)&sM1[wave][0];
            v4f acc = {0.f, 0.f, 0.f, 0.f};
            #pragma unroll
            for (int i = 0; i < ATT / 4; ++i) acc += wrow[i] * m1[i];
            const float o2 = d2b[lane] + acc[0] + acc[1] + acc[2] + acc[3];
            red = fmaxf(o2, 0.f) * fw[lane];
        }
        red += __shfl_xor(red, 8);
        red += __shfl_xor(red, 4);
        red += __shfl_xor(red, 2);
        red += __shfl_xor(red, 1);
        if (lane == 0)
            out[b] = 1.f / (1.f + __expf(-(red + fb[0])));
        __syncthreads();
    }
}

extern "C" void kernel_launch(void* const* d_in, const int* in_sizes, int n_in,
                              void* d_out, int out_size, void* d_ws, size_t ws_size,
                              hipStream_t stream) {
    const int*   x    = (const int*)  d_in[0];
    const float* emb  = (const float*)d_in[1];
    const float* W    = (const float*)d_in[2];
    const float* attb = (const float*)d_in[3];
    const float* hvec = (const float*)d_in[4];
    const float* d1w  = (const float*)d_in[5];
    const float* d1b  = (const float*)d_in[6];
    const float* d2w  = (const float*)d_in[7];
    const float* d2b  = (const float*)d_in[8];
    const float* fw   = (const float*)d_in[9];
    const float* fb   = (const float*)d_in[10];
    float* outp = (float*)d_out;

    // 128 blocks x 8 waves = 1024 waves; each wave processes 8 batch elements.
    afm_wmma_kernel<<<128, 256, 0, stream>>>(x, emb, W, attb, hvec,
                                             d1w, d1b, d2w, d2b, fw, fb, outp);
}
